// LSTM_52664888983991
// MI455X (gfx1250) — compile-verified, matrix operand order checked by reading, not run on hardware
//
#include <hip/hip_runtime.h>

// Problem dims
#define T_   512
#define B_   64
#define F_   256
#define H_   1024
#define C_   257
#define G4H  4096   // 4*H (f,i,o,c interleaved per unit: col = unit*4 + gate)
#define KTOT 1280   // F + H fused K
#define NWG  64     // persistent workgroups; each owns 16 units x 4 gates
#define CPAD 320    // C padded to 5 * 64

typedef __bf16        v16bf __attribute__((ext_vector_type(16)));
typedef float         v8f   __attribute__((ext_vector_type(8)));
typedef unsigned int  v4u   __attribute__((ext_vector_type(4)));
typedef unsigned int  u32x4 __attribute__((ext_vector_type(4)));
typedef int           i32x4 __attribute__((ext_vector_type(4)));
typedef int           i32x8 __attribute__((ext_vector_type(8)));

union Frag { v4u q[2]; v16bf v; };

__device__ inline unsigned short f2bf(float f) {
  unsigned u = __float_as_uint(f);
  return (unsigned short)((u + 0x7FFFu + ((u >> 16) & 1u)) >> 16);  // RNE
}
__device__ inline float sigf(float x) { return 1.0f / (1.0f + __expf(-x)); }
// branch-free tanh: 2*sigmoid(2x)-1  (v_exp_f32 + v_rcp_f32, no EXEC games)
__device__ inline float tanh_fast(float x) {
  return __builtin_fmaf(2.0f, 1.0f / (1.0f + __expf(-2.0f * x)), -1.0f);
}

// ---------------- prep kernels ----------------

__global__ void k_prep_x(const float* __restrict__ x, unsigned short* __restrict__ xb) {
  size_t i = ((size_t)blockIdx.x * blockDim.x + threadIdx.x) * 4;
#pragma unroll
  for (int j = 0; j < 4; ++j) xb[i + j] = f2bf(x[i + j]);
}

// Fused weight matrix Wc[4096][1280] bf16, gate-interleaved rows:
// row r = wg*64 + (unit_local*4 + gate); k<256 -> Wx[gate][unit][k], else Wh[gate][unit][k-256].
__global__ void k_prep_w(const float* __restrict__ wfx, const float* __restrict__ wix,
                         const float* __restrict__ wox, const float* __restrict__ wcx,
                         const float* __restrict__ bfx, const float* __restrict__ bix,
                         const float* __restrict__ box, const float* __restrict__ bcx,
                         const float* __restrict__ wfh, const float* __restrict__ wih,
                         const float* __restrict__ woh, const float* __restrict__ wch,
                         unsigned short* __restrict__ Wc, float* __restrict__ bc) {
  size_t idx = (size_t)blockIdx.x * blockDim.x + threadIdx.x;
  if (idx >= (size_t)G4H * KTOT) return;
  int r  = (int)(idx / KTOT), k = (int)(idx % KTOT);
  int wg = r >> 6, lc = r & 63, ul = lc >> 2, g = lc & 3;
  int u  = wg * 16 + ul;
  const float* Wx[4] = {wfx, wix, wox, wcx};
  const float* Wh[4] = {wfh, wih, woh, wch};
  float v = (k < F_) ? Wx[g][(size_t)u * F_ + k] : Wh[g][(size_t)u * H_ + (k - F_)];
  Wc[idx] = f2bf(v);
  if (k == 0) {
    const float* Bx[4] = {bfx, bix, box, bcx};
    bc[r] = Bx[g][u];
  }
}

__global__ void k_prep_fco(const float* __restrict__ fco, unsigned short* __restrict__ fb) {
  int idx = blockIdx.x * blockDim.x + threadIdx.x;   // CPAD*1024 threads
  int ch = idx >> 10, k = idx & 1023;
  float v = (ch < C_) ? fco[(size_t)ch * H_ + k] : 0.0f;
  fb[idx] = f2bf(v);
}

// ---------------- persistent LSTM recurrence ----------------
// 64 WGs x 256 threads (8 waves). Each WG: 64 output cols (16 units x 4 gates),
// weight slice (64 x 1280 bf16 = 160 KB) resident in LDS for all 512 steps.
__launch_bounds__(256, 1)
__global__ void k_lstm(const unsigned short* __restrict__ xb,   // [T][B][F] bf16
                       const unsigned short* __restrict__ Wc,   // [4096][1280] bf16
                       const float* __restrict__ bc,            // [4096]
                       unsigned short* __restrict__ hs,         // [(T+1)][B][H] bf16, row0=0
                       unsigned int* __restrict__ bar) {
  extern __shared__ char smem[];
  unsigned short* lds_w = (unsigned short*)smem;                        // 64*1280*2 = 163840 B
  float* lds_z = (float*)(smem + 64 * KTOT * 2);                        // 64*64*4   =  16384 B
  float* lds_c = (float*)(smem + 64 * KTOT * 2 + 64 * 64 * 4);          // 64*16*4   =   4096 B
  float* lds_b = (float*)(smem + 64 * KTOT * 2 + 64 * 64 * 4 + 64*16*4);// 64*4      =    256 B

  const int tid  = threadIdx.x;
  const int wg   = blockIdx.x;
  const int w    = tid >> 5;          // wave32
  const int lane = tid & 31;
  const int rb0  = 32 * (w >> 2);     // local batch-row base (acc0 rows, acc1 = +16)
  const int ncb  = 16 * (w & 3);      // local col base
  const int mrow = lane & 15;
  const int kk   = (lane >> 4) << 3;  // K sub-offset per ISA A/B fragment layout

  // ---- Stage this WG's weight slice into LDS (reused 512x) ----
#if __has_builtin(__builtin_amdgcn_tensor_load_to_lds) && __has_builtin(__builtin_amdgcn_s_wait_tensorcnt)
  if (w == 0) {
    // Build Tensor DMA Descriptor (D#), 1-D tile: 40960 dwords (the full slice),
    // data_size=4B, contiguous in global memory.  ISA 08_async_tensor §8.3/8.4.
    unsigned lds_off = (unsigned)(size_t)(__attribute__((address_space(3))) char*)smem;
    unsigned long long ga =
        (unsigned long long)(size_t)(Wc + (size_t)wg * 64 * KTOT);
    const unsigned ND = 40960u;                       // dwords in tile / tensor dim0
    u32x4 g0;
    g0[0] = 1u;                                       // count=1 (valid), user mode
    g0[1] = lds_off;                                  // lds_addr (bytes)
    g0[2] = (unsigned)(ga & 0xFFFFFFFFu);             // global_addr[31:0]
    g0[3] = (unsigned)((ga >> 32) & 0x01FFFFFFu)      // global_addr[56:32]
          | (2u << 30);                               // type=2 ("image")
    i32x8 g1;
    g1[0] = (int)(2u << 16);                          // wg_mask=0, data_size=2 (4B)
    g1[1] = (int)((ND & 0xFFFFu) << 16);              // tensor_dim0[15:0] in [63:48]
    g1[2] = (int)(((ND >> 16) & 0xFFFFu) | (1u << 16)); // tensor_dim0[31:16], tensor_dim1=1
    g1[3] = (int)((ND & 0xFFFFu) << 16);              // tensor_dim1 hi=0, tile_dim0=ND
    g1[4] = (int)(((ND >> 16) & 0xFFFFu) == 0 ? 1u : 1u); // tile_dim1=1, tile_dim2=0
    g1[5] = (int)ND;                                  // tensor_dim0_stride[31:0]
    g1[6] = 0;                                        // stride0[47:32], stride1[15:0]
    g1[7] = 0;
    i32x4 g2 = {0, 0, 0, 0};
    i32x4 g3 = {0, 0, 0, 0};
#if __has_include(<hip/amd_detail/amd_gfx1250_TDM.h>)
    i32x8 g4 = {0, 0, 0, 0, 0, 0, 0, 0};
    __builtin_amdgcn_tensor_load_to_lds(g0, g1, g2, g3, g4, 0);
#else
    __builtin_amdgcn_tensor_load_to_lds(g0, g1, g2, g3, 0);
#endif
  }
  __builtin_amdgcn_s_wait_tensorcnt(0);
#else
  {
    const v4u* src = (const v4u*)(Wc + (size_t)wg * 64 * KTOT);
    v4u* dst = (v4u*)lds_w;
    for (int i = tid; i < 64 * KTOT / 8; i += 256) dst[i] = src[i];
  }
#endif
  for (int i = tid; i < 64 * 16; i += 256) lds_c[i] = 0.0f;   // cell state c0 = 0
  if (tid < 64) lds_b[tid] = bc[wg * 64 + tid];               // per-WG biases
  __syncthreads();

  for (int t = 0; t < T_; ++t) {
    const unsigned short* Ax = xb + (size_t)t * B_ * F_;
    const unsigned short* Ah = hs + (size_t)t * B_ * H_;      // h_{t-1} (row 0 zeros)
    v8f acc0 = {}; v8f acc1 = {};
#pragma unroll 4
    for (int kc = 0; kc < KTOT; kc += 32) {
      Frag a0, a1, bfr;
      if (kc < F_) {   // uniform branch: K in x-projection region
        const unsigned short* p0 = Ax + (size_t)(rb0 + mrow) * F_ + kc + kk;
        const unsigned short* p1 = Ax + (size_t)(rb0 + 16 + mrow) * F_ + kc + kk;
        a0.q[0] = *(const v4u*)p0; a0.q[1] = *(const v4u*)(p0 + 16);
        a1.q[0] = *(const v4u*)p1; a1.q[1] = *(const v4u*)(p1 + 16);
      } else {         // K in recurrent region: read h_{t-1}
        const unsigned short* p0 = Ah + (size_t)(rb0 + mrow) * H_ + (kc - F_) + kk;
        const unsigned short* p1 = Ah + (size_t)(rb0 + 16 + mrow) * H_ + (kc - F_) + kk;
        a0.q[0] = *(const v4u*)p0; a0.q[1] = *(const v4u*)(p0 + 16);
        a1.q[0] = *(const v4u*)p1; a1.q[1] = *(const v4u*)(p1 + 16);
      }
      const unsigned short* qb = lds_w + (size_t)(ncb + mrow) * KTOT + kc + kk;
      bfr.q[0] = *(const v4u*)qb; bfr.q[1] = *(const v4u*)(qb + 16);
      acc0 = __builtin_amdgcn_wmma_f32_16x16x32_bf16(false, a0.v, false, bfr.v,
                                                     (short)0, acc0, false, false);
      acc1 = __builtin_amdgcn_wmma_f32_16x16x32_bf16(false, a1.v, false, bfr.v,
                                                     (short)0, acc1, false, false);
    }
    // Scatter accumulators to z tile (C/D layout: M = i + 8*(lane>>4), N = lane&15).
#pragma unroll
    for (int i = 0; i < 8; ++i) {
      int m = rb0 + i + ((lane >> 4) << 3);
      int n = ncb + (lane & 15);
      lds_z[m * 64 + n]        = acc0[i];
      lds_z[(m + 16) * 64 + n] = acc1[i];
    }
    __syncthreads();
    // Gate math: 64 batches x 16 local units = 1024 elems, 4 per thread. Branch-free.
#pragma unroll
    for (int p2 = 0; p2 < 4; ++p2) {
      int e = tid + p2 * 256;
      int b = e >> 4, ul = e & 15;
      float zf = lds_z[b * 64 + ul * 4 + 0] + lds_b[ul * 4 + 0];
      float zi = lds_z[b * 64 + ul * 4 + 1] + lds_b[ul * 4 + 1];
      float zo = lds_z[b * 64 + ul * 4 + 2] + lds_b[ul * 4 + 2];
      float za = lds_z[b * 64 + ul * 4 + 3] + lds_b[ul * 4 + 3];
      float c_old = lds_c[b * 16 + ul];
      float cn = sigf(zi) * tanh_fast(za) + sigf(zf) * c_old;
      float hn = sigf(zo) * tanh_fast(cn);
      lds_c[b * 16 + ul] = cn;
      hs[(size_t)(t + 1) * B_ * H_ + (size_t)b * H_ + wg * 16 + ul] = f2bf(hn);
    }
    // Grid-wide step barrier (monotonic counter; reset per launch via memset).
    __threadfence();
    __syncthreads();
    if (tid == 0) {
      __hip_atomic_fetch_add(bar, 1u, __ATOMIC_ACQ_REL, __HIP_MEMORY_SCOPE_AGENT);
      unsigned target = (unsigned)(NWG * (t + 1));
      while (__hip_atomic_load(bar, __ATOMIC_ACQUIRE, __HIP_MEMORY_SCOPE_AGENT) < target)
        __builtin_amdgcn_s_sleep(1);
    }
    __syncthreads();
  }
}

// ---------------- output projection: out[t,b,c] = hs @ fco^T + fco_b ----------------
__launch_bounds__(256)
__global__ void k_proj(const unsigned short* __restrict__ hsA,   // hs + B*H (skip h0 row)
                       const unsigned short* __restrict__ fcob,  // [CPAD][1024] bf16
                       const float* __restrict__ fco_b,
                       float* __restrict__ out) {
  const int tid  = threadIdx.x;
  const int w    = tid >> 5, lane = tid & 31;
  const int Mb   = blockIdx.x * 64 + 32 * (w >> 2);
  const int Nb   = blockIdx.y * 64 + 16 * (w & 3);
  const int mrow = lane & 15, kk = (lane >> 4) << 3;
  v8f acc0 = {}, acc1 = {};
  for (int kc = 0; kc < H_; kc += 32) {
    Frag a0, a1, bfr;
    const unsigned short* p0 = hsA + (size_t)(Mb + mrow) * H_ + kc + kk;
    const unsigned short* p1 = hsA + (size_t)(Mb + 16 + mrow) * H_ + kc + kk;
    __builtin_prefetch(p0 + 32, 0, 0);   // global_prefetch_b8 next K chunk
    a0.q[0] = *(const v4u*)p0; a0.q[1] = *(const v4u*)(p0 + 16);
    a1.q[0] = *(const v4u*)p1; a1.q[1] = *(const v4u*)(p1 + 16);
    const unsigned short* qb = fcob + (size_t)(Nb + mrow) * H_ + kc + kk;
    bfr.q[0] = *(const v4u*)qb; bfr.q[1] = *(const v4u*)(qb + 16);
    acc0 = __builtin_amdgcn_wmma_f32_16x16x32_bf16(false, a0.v, false, bfr.v,
                                                   (short)0, acc0, false, false);
    acc1 = __builtin_amdgcn_wmma_f32_16x16x32_bf16(false, a1.v, false, bfr.v,
                                                   (short)0, acc1, false, false);
  }
#pragma unroll
  for (int i = 0; i < 8; ++i) {
    int m  = Mb + i + ((lane >> 4) << 3);
    int ch = Nb + (lane & 15);
    if (ch < C_) {
      float bb = fco_b[ch];
      out[(size_t)m * C_ + ch]        = acc0[i] + bb;
      out[(size_t)(m + 16) * C_ + ch] = acc1[i] + bb;
    }
  }
}

// ---------------- launcher ----------------
extern "C" void kernel_launch(void* const* d_in, const int* in_sizes, int n_in,
                              void* d_out, int out_size, void* d_ws, size_t ws_size,
                              hipStream_t stream) {
  (void)in_sizes; (void)n_in; (void)out_size; (void)ws_size;
  const float* x     = (const float*)d_in[0];
  const float* wfx_w = (const float*)d_in[1];  const float* wfx_b = (const float*)d_in[2];
  const float* wix_w = (const float*)d_in[3];  const float* wix_b = (const float*)d_in[4];
  const float* wox_w = (const float*)d_in[5];  const float* wox_b = (const float*)d_in[6];
  const float* wcx_w = (const float*)d_in[7];  const float* wcx_b = (const float*)d_in[8];
  const float* wfh_w = (const float*)d_in[9];
  const float* wih_w = (const float*)d_in[10];
  const float* woh_w = (const float*)d_in[11];
  const float* wch_w = (const float*)d_in[12];
  const float* fco_w = (const float*)d_in[13];
  const float* fco_b = (const float*)d_in[14];

  // Workspace layout (all 256 B aligned)
  char* ws = (char*)d_ws;
  unsigned int* bar   = (unsigned int*)ws;                      size_t off = 256;
  unsigned short* xb  = (unsigned short*)(ws + off);            off += (size_t)T_ * B_ * F_ * 2;    // 16 MB
  unsigned short* Wc  = (unsigned short*)(ws + off);            off += (size_t)G4H * KTOT * 2;       // 10 MB
  float* bc           = (float*)(ws + off);                     off += (size_t)G4H * 4;
  unsigned short* fcb = (unsigned short*)(ws + off);            off += (size_t)CPAD * H_ * 2;
  off = (off + 255) & ~(size_t)255;
  unsigned short* hsb = (unsigned short*)(ws + off);            // (T+1)*B*H bf16 = 64 MB

  hipMemsetAsync(bar, 0, 256, stream);                          // reset grid barrier
  hipMemsetAsync(hsb, 0, (size_t)B_ * H_ * 2, stream);          // h0 = 0

  k_prep_x<<<(T_ * B_ * F_) / 4 / 256, 256, 0, stream>>>(x, xb);
  k_prep_w<<<((size_t)G4H * KTOT + 255) / 256, 256, 0, stream>>>(
      wfx_w, wix_w, wox_w, wcx_w, wfx_b, wix_b, wox_b, wcx_b,
      wfh_w, wih_w, woh_w, wch_w, Wc, bc);
  k_prep_fco<<<(CPAD * H_) / 256, 256, 0, stream>>>(fco_w, fcb);

  const size_t lds_bytes = 64 * KTOT * 2 + 64 * 64 * 4 + 64 * 16 * 4 + 64 * 4; // 184576 B
  k_lstm<<<NWG, 256, lds_bytes, stream>>>(xb, Wc, bc, hsb, bar);

  dim3 gproj((T_ * B_) / 64, (C_ + 63) / 64);   // 512 x 5
  k_proj<<<gproj, 256, 0, stream>>>(hsb + (size_t)B_ * H_, fcb, fco_b, (float*)d_out);
}
